// BlockMoE_82961588290175
// MI455X (gfx1250) — matrix operand order repaired and checked by reading, not compiled, not run on hardware
//
#include <hip/hip_runtime.h>

typedef unsigned short ushort_t;
typedef __attribute__((ext_vector_type(16))) __bf16 bf16x16;
typedef __attribute__((ext_vector_type(8)))  float  f32x8;

union FragB { bf16x16 v; uint4 q[2]; };

__device__ __forceinline__ ushort_t f2bf(float f) {
    union { float f; unsigned u; } c; c.f = f;
    unsigned u = c.u;
    unsigned r = u + 0x7FFFu + ((u >> 16) & 1u);   // RNE
    return (ushort_t)(r >> 16);
}
__device__ __forceinline__ float bf2f(ushort_t h) {
    union { unsigned u; float f; } c; c.u = ((unsigned)h) << 16;
    return c.f;
}
__device__ __forceinline__ float gelu_exact(float v) {
    return 0.5f * v * (1.0f + erff(v * 0.70710678118654752f));
}

// ---------------- problem constants ----------------
#define BB   8
#define TT   197
#define DD   768
#define NTOK 1576          // B*T
#define HHD  12            // heads
#define HD   64            // head dim
#define EE   8             // experts
#define HID  3072          // expert hidden
#define QKVC 2304
#define MT   64
#define NT   128
#define KT   32
#define MAXTILES 64
#define SLOT_CAP 4096      // 64 tiles * 64 rows

// ---------------- weight transpose fp32 -> bf16, [Bt,K,Nc] -> [Bt,Nc,K] ------------
__global__ void transpose_cvt_kernel(const float* __restrict__ in, ushort_t* __restrict__ out,
                                     int K, int Nc) {
    __shared__ float tile[32][33];
    int b = blockIdx.z;
    const float* I = in + (size_t)b * K * Nc;
    ushort_t* O = out + (size_t)b * K * Nc;
    int n0 = blockIdx.x * 32, k0 = blockIdx.y * 32;
    int tx = threadIdx.x, ty = threadIdx.y;
    #pragma unroll
    for (int i = 0; i < 32; i += 8)
        tile[ty + i][tx] = I[(size_t)(k0 + ty + i) * Nc + n0 + tx];
    __syncthreads();
    #pragma unroll
    for (int i = 0; i < 32; i += 8)
        O[(size_t)(n0 + ty + i) * K + k0 + tx] = f2bf(tile[tx][ty + i]);
}

// ---------------- tiny zero-fill for async-copy padding source ----------------
__global__ void zero_kernel(ushort_t* __restrict__ z) { z[threadIdx.x] = 0; }

// ---------------- layernorm fp32 -> bf16, one block per row (D=768) ----------------
__global__ __launch_bounds__(256) void ln_kernel(const float* __restrict__ x,
                                                 const float* __restrict__ w,
                                                 const float* __restrict__ b,
                                                 ushort_t* __restrict__ out) {
    __shared__ float red[256];
    int row = blockIdx.x, tid = threadIdx.x;
    const float* xr = x + (size_t)row * DD;
    float v0 = xr[tid], v1 = xr[tid + 256], v2 = xr[tid + 512];
    red[tid] = v0 + v1 + v2;
    __syncthreads();
    for (int o = 128; o; o >>= 1) { if (tid < o) red[tid] += red[tid + o]; __syncthreads(); }
    float mu = red[0] * (1.0f / DD);
    __syncthreads();
    float d0 = v0 - mu, d1 = v1 - mu, d2 = v2 - mu;
    red[tid] = d0 * d0 + d1 * d1 + d2 * d2;
    __syncthreads();
    for (int o = 128; o; o >>= 1) { if (tid < o) red[tid] += red[tid + o]; __syncthreads(); }
    float rs = rsqrtf(red[0] * (1.0f / DD) + 1e-5f);
    ushort_t* orow = out + (size_t)row * DD;
    orow[tid]       = f2bf(d0 * rs * w[tid]       + b[tid]);
    orow[tid + 256] = f2bf(d1 * rs * w[tid + 256] + b[tid + 256]);
    orow[tid + 512] = f2bf(d2 * rs * w[tid + 512] + b[tid + 512]);
}

// ---------------- WMMA GEMM: C[M,Nc] = A[M,K](bf16) x W^T[Nc,K](bf16) --------------
// Global->LDS tiles move via GLOBAL_LOAD_ASYNC_TO_LDS_B128 (ASYNCcnt), double buffered.
// MODE 0: store bf16 (qkv)
// MODE 1: +residual fp32, store fp32 (proj)
// MODE 2: grouped, gather A rows via row_token, +bias, gelu, store bf16 (MoE up)
// MODE 3: grouped, A = h by slot, +bias, store fp32 (MoE down)
template <int MODE>
__global__ __launch_bounds__(256) void gemm_wmma_kernel(
    const ushort_t* __restrict__ Ab, const ushort_t* __restrict__ Wt,
    ushort_t* __restrict__ outB, float* __restrict__ outF,
    const float* __restrict__ bias, const float* __restrict__ resid,
    const int* __restrict__ row_token, const int* __restrict__ tile_expert,
    const ushort_t* __restrict__ zbuf,
    int M, int K, int Nc)
{
    __shared__ alignas(16) ushort_t As[2][MT][40];
    __shared__ alignas(16) ushort_t Bs[2][NT][40];

    int tid = threadIdx.x;
    int mtile = blockIdx.y, ntile = blockIdx.x;
    int expert = 0;
    if (MODE >= 2) { expert = tile_expert[mtile]; if (expert < 0) return; }
    int mbase = mtile * MT, nbase = ntile * NT;
    const ushort_t* Wrow = Wt + (size_t)expert * K * Nc + (size_t)nbase * K;

    // cooperative loaders
    int arow = tid >> 2, ach = tid & 3;     // A: 64 rows x (4 chunks * 8 elems = 16B)
    int brow = tid >> 1, bh2 = tid & 1;     // B: 128 rows x (2 halves * 16 elems = 32B)

    size_t aoff = 0; bool avalid = true;
    if (MODE == 2) {
        int tok = row_token[mbase + arow];
        avalid = (tok >= 0);
        aoff = (size_t)(avalid ? tok : 0) * K;
    } else if (MODE == 3) {
        aoff = (size_t)(mbase + arow) * K;
    } else {
        int r = mbase + arow;
        avalid = (r < M);
        aoff = (size_t)(avalid ? r : 0) * K;
    }
    const ushort_t* Ap = Ab + aoff + ach * 8;
    const ushort_t* Bp = Wrow + (size_t)brow * K + bh2 * 16;
    int KS = K / KT;

    // async global->LDS tile copy: 16B for A, 32B for B per thread
    auto issue_tile = [&](int ks, int buf) {
        const ushort_t* ap = avalid ? (Ap + (size_t)ks * KT) : zbuf;
        const ushort_t* bp = Bp + (size_t)ks * KT;
        unsigned la = (unsigned)(size_t)&As[buf][arow][ach * 8];
        unsigned lb = (unsigned)(size_t)&Bs[buf][brow][bh2 * 16];
        asm volatile("global_load_async_to_lds_b128 %0, %1, off"
                     :: "v"(la), "v"(ap) : "memory");
        asm volatile("global_load_async_to_lds_b128 %0, %1, off"
                     :: "v"(lb), "v"(bp) : "memory");
        asm volatile("global_load_async_to_lds_b128 %0, %1, off offset:16"
                     :: "v"(lb), "v"(bp) : "memory");   // INST_OFFSET applies to LDS and global
    };

    issue_tile(0, 0);
    asm volatile("s_wait_asynccnt 0x0" ::: "memory");
    __syncthreads();

    int lane = tid & 31, wv = tid >> 5;
    int wm = (wv & 3) * 16;     // wave M offset in tile
    int wn = (wv >> 2) * 64;    // wave N offset in tile

    f32x8 acc[4];
    #pragma unroll
    for (int i = 0; i < 4; i++)
        #pragma unroll
        for (int e = 0; e < 8; e++) acc[i][e] = 0.0f;

    int arl   = wm + (lane & 15);
    int akoff = (lane >> 4) * 8;    // A frag: K [akoff,akoff+8) and [akoff+16,akoff+24)
    int bkoff = (lane >> 4) * 8;    // B frag: K [bkoff,bkoff+8) and [bkoff+16,bkoff+24)

    for (int ks = 0; ks < KS; ks++) {
        int cur = ks & 1;
        if (ks + 1 < KS) issue_tile(ks + 1, cur ^ 1);   // overlaps with compute below
        FragB fa;
        fa.q[0] = *(const uint4*)&As[cur][arl][akoff];
        fa.q[1] = *(const uint4*)&As[cur][arl][akoff + 16];
        #pragma unroll
        for (int i = 0; i < 4; i++) {
            FragB fb;
            const ushort_t* bp = &Bs[cur][wn + i * 16 + (lane & 15)][bkoff];
            fb.q[0] = *(const uint4*)bp;
            fb.q[1] = *(const uint4*)(bp + 16);
            acc[i] = __builtin_amdgcn_wmma_f32_16x16x32_bf16(
                false, fa.v, false, fb.v, (short)0, acc[i], false, false);
        }
        if (ks + 1 < KS) {
            asm volatile("s_wait_asynccnt 0x0" ::: "memory");
            __syncthreads();
        }
    }

    // epilogue: C 16x16 layout — lane holds col (lane&15); vgpr e => row e (+8 for lanes>=16)
    int rl0 = wm + ((lane >> 4) << 3);
    int cl  = lane & 15;
    #pragma unroll
    for (int i = 0; i < 4; i++) {
        int col = nbase + wn + i * 16 + cl;
        #pragma unroll
        for (int e = 0; e < 8; e++) {
            int grow = mbase + rl0 + e;
            float v = acc[i][e];
            if (MODE == 0) {
                if (grow < M) outB[(size_t)grow * Nc + col] = f2bf(v);
            } else if (MODE == 1) {
                if (grow < M) {
                    size_t o = (size_t)grow * Nc + col;
                    outF[o] = v + resid[o];
                }
            } else if (MODE == 2) {
                v += bias[expert * Nc + col];
                outB[(size_t)grow * Nc + col] = f2bf(gelu_exact(v));
            } else {
                v += bias[expert * Nc + col];
                outF[(size_t)grow * Nc + col] = v;
            }
        }
    }
}

// ---------------- qkv reshape: [N,2304] -> q/k [BH,T,64] bf16, v^T [BH,64,T] ----------
__global__ void qkv_reshape_kernel(const ushort_t* __restrict__ qkv,
                                   ushort_t* __restrict__ q, ushort_t* __restrict__ k,
                                   ushort_t* __restrict__ vt) {
    int idx = blockIdx.x * 256 + threadIdx.x;   // < NTOK*QKVC exactly
    int n = idx / QKVC, c = idx % QKVC;
    int cc = c / DD, rem = c % DD, h = rem >> 6, d = rem & 63;
    int b = n / TT, t = n % TT;
    int bh = b * HHD + h;
    ushort_t val = qkv[idx];
    if (cc == 0)       q[((size_t)bh * TT + t) * HD + d] = val;
    else if (cc == 1)  k[((size_t)bh * TT + t) * HD + d] = val;
    else               vt[(size_t)bh * HD * TT + (size_t)d * TT + t] = val;
}

// ---------------- attention core: one wave per (b,h,q-row) ----------------
__global__ __launch_bounds__(256) void attn_kernel(const ushort_t* __restrict__ q,
                                                   const ushort_t* __restrict__ k,
                                                   const ushort_t* __restrict__ vt,
                                                   ushort_t* __restrict__ a) {
    __shared__ float qs[8][64];
    __shared__ float sc[8][200];
    int tid = threadIdx.x, lane = tid & 31, w = tid >> 5;
    int bh = blockIdx.x;
    int qr = blockIdx.y * 8 + w;
    bool act = (qr < TT);
    if (act) {
        const ushort_t* qrow = q + ((size_t)bh * TT + qr) * HD;
        qs[w][lane * 2]     = bf2f(qrow[lane * 2]);
        qs[w][lane * 2 + 1] = bf2f(qrow[lane * 2 + 1]);
    }
    float mloc = -1e30f;
    if (act) {
        for (int j = lane; j < TT; j += 32) {
            const ushort_t* kr = k + ((size_t)bh * TT + j) * HD;
            float s = 0.0f;
            #pragma unroll 8
            for (int d = 0; d < HD; d++) s += qs[w][d] * bf2f(kr[d]);
            s *= 0.125f;            // hd^-0.5
            sc[w][j] = s;
            mloc = fmaxf(mloc, s);
        }
    }
    #pragma unroll
    for (int o = 16; o; o >>= 1) mloc = fmaxf(mloc, __shfl_xor(mloc, o, 32));
    float ssum = 0.0f;
    if (act) {
        for (int j = lane; j < TT; j += 32) {
            float p = __expf(sc[w][j] - mloc);
            sc[w][j] = p;
            ssum += p;
        }
    }
    #pragma unroll
    for (int o = 16; o; o >>= 1) ssum += __shfl_xor(ssum, o, 32);
    if (act) {
        float inv = 1.0f / ssum;
        int d0 = lane * 2;
        const ushort_t* v0r = vt + (size_t)bh * HD * TT + (size_t)d0 * TT;
        const ushort_t* v1r = v0r + TT;
        float a0 = 0.0f, a1 = 0.0f;
        for (int j = 0; j < TT; j++) {
            float p = sc[w][j];
            a0 += p * bf2f(v0r[j]);
            a1 += p * bf2f(v1r[j]);
        }
        int b = bh / HHD, h = bh % HHD;
        size_t o = (size_t)(b * TT + qr) * DD + h * HD + d0;
        a[o]     = f2bf(a0 * inv);
        a[o + 1] = f2bf(a1 * inv);
    }
}

// ---------------- router: one wave per token -> probs, lse, top2 ----------------
__global__ __launch_bounds__(256) void router_kernel(const ushort_t* __restrict__ xn2,
                                                     const float* __restrict__ Wg,
                                                     float* __restrict__ probs,
                                                     float* __restrict__ lse,
                                                     int* __restrict__ topi,
                                                     float* __restrict__ topv) {
    int tid = threadIdx.x, lane = tid & 31, w = tid >> 5;
    int n = blockIdx.x * 8 + w;
    float acc[EE];
    #pragma unroll
    for (int e = 0; e < EE; e++) acc[e] = 0.0f;
    if (n < NTOK) {
        const ushort_t* xr = xn2 + (size_t)n * DD;
        for (int j = lane; j < DD; j += 32) {
            float xv = bf2f(xr[j]);
            const float* wg = Wg + (size_t)j * EE;
            #pragma unroll
            for (int e = 0; e < EE; e++) acc[e] += xv * wg[e];
        }
    }
    #pragma unroll
    for (int e = 0; e < EE; e++)
        #pragma unroll
        for (int o = 16; o; o >>= 1) acc[e] += __shfl_xor(acc[e], o, 32);
    if (n < NTOK && lane == 0) {
        float mx = acc[0];
        #pragma unroll
        for (int e = 1; e < EE; e++) mx = fmaxf(mx, acc[e]);
        float p[EE], s = 0.0f;
        #pragma unroll
        for (int e = 0; e < EE; e++) { p[e] = __expf(acc[e] - mx); s += p[e]; }
        float inv = 1.0f / s;
        #pragma unroll
        for (int e = 0; e < EE; e++) { p[e] *= inv; probs[(size_t)n * EE + e] = p[e]; }
        lse[n] = logf(s) + mx;
        int i0 = 0; float v0 = p[0];
        #pragma unroll
        for (int e = 1; e < EE; e++) if (p[e] > v0) { v0 = p[e]; i0 = e; }
        int i1 = -1; float v1 = -1.0f;
        #pragma unroll
        for (int e = 0; e < EE; e++) if (e != i0 && p[e] > v1) { v1 = p[e]; i1 = e; }
        topi[n * 2] = i0; topi[n * 2 + 1] = i1;
        topv[n * 2] = v0; topv[n * 2 + 1] = v1;
    }
}

// ---------------- routing finalize (single block): sort slots, tiles, aux losses -----
__global__ __launch_bounds__(256) void route_finalize_kernel(
    const int* __restrict__ topi, const float* __restrict__ probs,
    const float* __restrict__ lse, int* __restrict__ slot_of,
    int* __restrict__ row_token, int* __restrict__ tile_expert,
    float* __restrict__ out_scalars) {
    __shared__ int cnt[EE], base[EE], pad[EE], cnt2[EE];
    __shared__ float red[256];
    int tid = threadIdx.x;
    if (tid < EE) { cnt[tid] = 0; cnt2[tid] = 0; }
    for (int s = tid; s < SLOT_CAP; s += 256) row_token[s] = -1;
    __syncthreads();
    for (int i = tid; i < NTOK * 2; i += 256) atomicAdd(&cnt[topi[i]], 1);
    __syncthreads();
    if (tid == 0) {
        int off = 0, t = 0;
        for (int e = 0; e < EE; e++) {
            base[e] = off;
            pad[e] = (cnt[e] + MT - 1) & ~(MT - 1);
            off += pad[e];
        }
        for (int e = 0; e < EE; e++)
            for (int j = 0; j < pad[e] / MT; j++) tile_expert[t++] = e;
        for (; t < MAXTILES; t++) tile_expert[t] = -1;
    }
    __syncthreads();
    for (int i = tid; i < NTOK * 2; i += 256) {
        int e = topi[i];
        int pos = atomicAdd(&cnt2[e], 1);
        int s = base[e] + pos;
        slot_of[i] = s;
        row_token[s] = i >> 1;
    }
    // aux losses (fixed-order tree reductions -> deterministic)
    float vals[17];
    #pragma unroll
    for (int v = 0; v < 17; v++) vals[v] = 0.0f;
    for (int r = tid; r < NTOK; r += 256) {
        #pragma unroll
        for (int e = 0; e < EE; e++) vals[e] += probs[(size_t)r * EE + e];
        vals[EE + topi[r * 2]] += 1.0f;
        float l = lse[r];
        vals[16] += l * l;
    }
    float tot[17];
    for (int v = 0; v < 17; v++) {
        red[tid] = vals[v];
        __syncthreads();
        for (int o = 128; o; o >>= 1) { if (tid < o) red[tid] += red[tid + o]; __syncthreads(); }
        if (tid == 0) tot[v] = red[0];
        __syncthreads();
    }
    if (tid == 0) {
        float bal = 0.0f;
        for (int e = 0; e < EE; e++)
            bal += (tot[EE + e] / (float)NTOK) * (tot[e] / (float)NTOK);
        bal *= (float)EE;
        float z = tot[16] / (float)NTOK;
        out_scalars[0] = 0.01f * bal + 0.001f * z;
        out_scalars[1] = bal;
        out_scalars[2] = z;
    }
}

// ---------------- combine: out[n] = 2*(w0*y[s0] + w1*y[s1]) ----------------
__global__ void combine_kernel(const float* __restrict__ y, const int* __restrict__ slot_of,
                               const float* __restrict__ topv, float* __restrict__ out) {
    int n = blockIdx.x;
    int d = blockIdx.y * 256 + threadIdx.x;
    int s0 = slot_of[n * 2], s1 = slot_of[n * 2 + 1];
    float v0 = topv[n * 2], v1 = topv[n * 2 + 1];
    out[(size_t)n * DD + d] = 2.0f * (v0 * y[(size_t)s0 * DD + d] + v1 * y[(size_t)s1 * DD + d]);
}

// ---------------- workspace layout (bytes) ----------------
#define WQKVT_OFF   0u
#define WPROJT_OFF  3538944u
#define W1T_OFF     4718592u
#define W2T_OFF     42467328u
#define XN_OFF      80216064u
#define QKV_OFF     82636800u
#define QB_OFF      89899008u
#define KB_OFF      92319744u
#define VT_OFF      94740480u
#define AB_OFF      97161216u
#define X1_OFF      99581952u
#define XN2_OFF     104423424u
#define H_OFF       106844160u
#define Y_OFF       132009984u
#define PROB_OFF    144592896u
#define LSE_OFF     144643328u
#define TOPI_OFF    144649632u
#define TOPV_OFF    144662240u
#define SLOT_OFF    144674848u
#define ROWTOK_OFF  144687456u
#define TILEE_OFF   144703840u
#define ZERO_OFF    144704096u

extern "C" void kernel_launch(void* const* d_in, const int* in_sizes, int n_in,
                              void* d_out, int out_size, void* d_ws, size_t ws_size,
                              hipStream_t stream) {
    (void)in_sizes; (void)n_in; (void)out_size; (void)ws_size;
    const float* x      = (const float*)d_in[0];
    const float* n1w    = (const float*)d_in[1];
    const float* n1b    = (const float*)d_in[2];
    const float* Wqkv   = (const float*)d_in[3];
    const float* Wproj  = (const float*)d_in[4];
    const float* n2w    = (const float*)d_in[5];
    const float* n2b    = (const float*)d_in[6];
    const float* Wg     = (const float*)d_in[7];
    const float* W1     = (const float*)d_in[8];
    const float* b1     = (const float*)d_in[9];
    const float* W2     = (const float*)d_in[10];
    const float* b2     = (const float*)d_in[11];
    float* out = (float*)d_out;

    char* ws = (char*)d_ws;
    ushort_t* wqkvT = (ushort_t*)(ws + WQKVT_OFF);
    ushort_t* wprojT= (ushort_t*)(ws + WPROJT_OFF);
    ushort_t* w1T   = (ushort_t*)(ws + W1T_OFF);
    ushort_t* w2T   = (ushort_t*)(ws + W2T_OFF);
    ushort_t* xn    = (ushort_t*)(ws + XN_OFF);
    ushort_t* qkv   = (ushort_t*)(ws + QKV_OFF);
    ushort_t* qb    = (ushort_t*)(ws + QB_OFF);
    ushort_t* kb    = (ushort_t*)(ws + KB_OFF);
    ushort_t* vtb   = (ushort_t*)(ws + VT_OFF);
    ushort_t* ab    = (ushort_t*)(ws + AB_OFF);
    float*    x1    = (float*)(ws + X1_OFF);
    ushort_t* xn2   = (ushort_t*)(ws + XN2_OFF);
    ushort_t* hbuf  = (ushort_t*)(ws + H_OFF);
    float*    ybuf  = (float*)(ws + Y_OFF);
    float*    probs = (float*)(ws + PROB_OFF);
    float*    lse   = (float*)(ws + LSE_OFF);
    int*      topi  = (int*)(ws + TOPI_OFF);
    float*    topv  = (float*)(ws + TOPV_OFF);
    int*      slot_of   = (int*)(ws + SLOT_OFF);
    int*      row_token = (int*)(ws + ROWTOK_OFF);
    int*      tile_exp  = (int*)(ws + TILEE_OFF);
    ushort_t* zbuf  = (ushort_t*)(ws + ZERO_OFF);

    dim3 tb(32, 8);
    zero_kernel<<<1, 64, 0, stream>>>(zbuf);
    // weight conversions (fp32 -> bf16 transposed)
    transpose_cvt_kernel<<<dim3(QKVC / 32, DD / 32, 1), tb, 0, stream>>>(Wqkv, wqkvT, DD, QKVC);
    transpose_cvt_kernel<<<dim3(DD / 32, DD / 32, 1),   tb, 0, stream>>>(Wproj, wprojT, DD, DD);
    transpose_cvt_kernel<<<dim3(HID / 32, DD / 32, EE), tb, 0, stream>>>(W1, w1T, DD, HID);
    transpose_cvt_kernel<<<dim3(DD / 32, HID / 32, EE), tb, 0, stream>>>(W2, w2T, HID, DD);

    // attention branch
    ln_kernel<<<NTOK, 256, 0, stream>>>(x, n1w, n1b, xn);
    gemm_wmma_kernel<0><<<dim3(QKVC / NT, (NTOK + MT - 1) / MT), 256, 0, stream>>>(
        xn, wqkvT, qkv, nullptr, nullptr, nullptr, nullptr, nullptr, zbuf, NTOK, DD, QKVC);
    qkv_reshape_kernel<<<(NTOK * QKVC) / 256, 256, 0, stream>>>(qkv, qb, kb, vtb);
    attn_kernel<<<dim3(BB * HHD, (TT + 7) / 8), 256, 0, stream>>>(qb, kb, vtb, ab);
    gemm_wmma_kernel<1><<<dim3(DD / NT, (NTOK + MT - 1) / MT), 256, 0, stream>>>(
        ab, wprojT, nullptr, x1, nullptr, x, nullptr, nullptr, zbuf, NTOK, DD, DD);

    // MoE branch
    ln_kernel<<<NTOK, 256, 0, stream>>>(x1, n2w, n2b, xn2);
    router_kernel<<<(NTOK + 7) / 8, 256, 0, stream>>>(xn2, Wg, probs, lse, topi, topv);
    route_finalize_kernel<<<1, 256, 0, stream>>>(topi, probs, lse, slot_of, row_token,
                                                 tile_exp, out + (size_t)NTOK * DD);
    gemm_wmma_kernel<2><<<dim3(HID / NT, MAXTILES), 256, 0, stream>>>(
        xn2, w1T, hbuf, nullptr, b1, nullptr, row_token, tile_exp, zbuf, 0, DD, HID);
    gemm_wmma_kernel<3><<<dim3(DD / NT, MAXTILES), 256, 0, stream>>>(
        hbuf, w2T, nullptr, ybuf, b2, nullptr, nullptr, tile_exp, zbuf, 0, HID, DD);
    combine_kernel<<<dim3(NTOK, DD / 256), 256, 0, stream>>>(ybuf, slot_of, topv, out);
}
